// EnDeModel_42880953483646
// MI455X (gfx1250) — compile-verified, matrix operand order checked by reading, not run on hardware
//
#include <hip/hip_runtime.h>

typedef __attribute__((ext_vector_type(16))) _Float16 v16h;
typedef __attribute__((ext_vector_type(8)))  _Float16 v8h;
typedef __attribute__((ext_vector_type(8)))  float    v8f;
typedef __attribute__((ext_vector_type(4)))  unsigned int v4u;
typedef __attribute__((ext_vector_type(8)))  int v8i;
typedef __attribute__((ext_vector_type(4)))  int v4i;

#define HIST 60
#define HDIM 128
#define TSTEPS 365
#define BATCH 4096
#define G3 384

// LDS layout (bytes). Total 267776 < 320KB/WGP -> dynamic LDS.
#define SM_W     0         // 48 n-tiles * 4 k-tiles * 1024B = 196608 (f16 B-fragments)
#define SM_BIH   196608    // 384 f32
#define SM_BHH   198144    // 384 f32
#define SM_BETA  199680    // [2 buf][2 half][128] f32 = 2048
#define SM_OW    201728    // 128 f32
#define SM_STAGE 202240    // double-buffered h: 2 * (128 rows * 128 u * f16) = 65536
#define SM_TOTAL 267776

#if defined(__has_builtin)
#if __has_builtin(__builtin_amdgcn_tensor_load_to_lds)
#define HAVE_TDM 1
#endif
#endif

__device__ __forceinline__ float fast_sigmoid(float x) {
  return __builtin_amdgcn_rcpf(1.0f + __expf(-x));
}
__device__ __forceinline__ float fast_tanh(float x) {
  return 1.0f - 2.0f * __builtin_amdgcn_rcpf(__expf(2.0f * x) + 1.0f);
}

// acc[mi] = bias + sum_kt A[kt][mi] x B(ntI,kt)  [+ sum_kt A[kt][mi] x B(ntH,kt)]
// All 4 B tiles are loaded before the MMA chain so the scheduler can stagger
// s_wait_dscnt instead of a full wait per WMMA pair.
__device__ __forceinline__ void gate_mma_pair(v8f acc[2], const v16h (&a)[4][2],
                                              const _Float16* wl, int ntI, int ntH,
                                              int lane, float bias, bool withH) {
  v16h b[4];
#pragma unroll
  for (int kt = 0; kt < 4; ++kt)
    b[kt] = *(const v16h*)(wl + (ntI * 4 + kt) * 512 + lane * 16);
#pragma unroll
  for (int mi = 0; mi < 2; ++mi)
#pragma unroll
    for (int i = 0; i < 8; ++i) acc[mi][i] = bias;
#pragma unroll
  for (int kt = 0; kt < 4; ++kt)
#pragma unroll
    for (int mi = 0; mi < 2; ++mi)
      acc[mi] = __builtin_amdgcn_wmma_f32_16x16x32_f16(
          false, a[kt][mi], false, b[kt], (short)0, acc[mi], false, false);
  if (withH) {
#pragma unroll
    for (int kt = 0; kt < 4; ++kt)
      b[kt] = *(const v16h*)(wl + (ntH * 4 + kt) * 512 + lane * 16);
#pragma unroll
    for (int kt = 0; kt < 4; ++kt)
#pragma unroll
      for (int mi = 0; mi < 2; ++mi)
        acc[mi] = __builtin_amdgcn_wmma_f32_16x16x32_f16(
            false, a[kt][mi], false, b[kt], (short)0, acc[mi], false, false);
  }
}

// ---------------- encoder: y0 = sigmoid(hist @ enc_w.T + enc_b) -> f16 ------
__global__ void __launch_bounds__(256)
enc_kernel_42880953483646(const float* __restrict__ hist,
                          const float* __restrict__ enc_w,
                          const float* __restrict__ enc_b,
                          _Float16* __restrict__ y0) {
  int idx = blockIdx.x * blockDim.x + threadIdx.x;  // b*128 + h
  int b = idx >> 7;
  int h = idx & 127;
  const float* hr = hist + b * HIST;
  const float* wr = enc_w + h * HIST;
  float acc = enc_b[h];
#pragma unroll 4
  for (int i = 0; i < HIST; ++i) acc = fmaf(hr[i], wr[i], acc);
  y0[idx] = (_Float16)fast_sigmoid(acc);
}

// ---------------- fused GRU decoder (365 steps) + beta + SIRD rollout -------
// 8 waves / block: slab = wave&3 owns 32 batch rows (2 M-tiles); half = wave>>2
// splits the 8 u-tiles (gates are elementwise in u). Double-buffered h stage
// -> exactly one barrier per step.
__global__ void __launch_bounds__(256)
gru_sird_kernel_42880953483646(const _Float16* __restrict__ y0g,
                               const float* __restrict__ sird,
                               const float* __restrict__ w_ih,
                               const float* __restrict__ w_hh,
                               const float* __restrict__ b_ih,
                               const float* __restrict__ b_hh,
                               const float* __restrict__ out_w,
                               const float* __restrict__ out_b,
                               float* __restrict__ out) {
  extern __shared__ char smem[];
  _Float16* wl    = (_Float16*)(smem + SM_W);
  float*    bihL  = (float*)(smem + SM_BIH);
  float*    bhhL  = (float*)(smem + SM_BHH);
  float*    betaL = (float*)(smem + SM_BETA);   // [buf][half][128]
  float*    owL   = (float*)(smem + SM_OW);

  const int tid  = threadIdx.x;
  const int lane = tid & 31;
  const int wave = tid >> 5;
  const int slab = wave & 3;    // which 32-row slab
  const int half = wave >> 2;   // which half of the u-tiles
  const int ln16 = lane & 15;
  const int g    = lane >> 4;
  const int blockRow0 = blockIdx.x * 128;
  const int slabRow0  = slab * 32;

#ifdef HAVE_TDM
  // ---- y0 tile (32KB f16) via Tensor Data Mover, overlapped with weight
  // staging below. 1-D descriptor: data_size=2B, tile_dim0 = 128*128 elems.
  if (wave == 0) {
    unsigned long long ga = (unsigned long long)(y0g + (size_t)blockRow0 * HDIM);
    v4u g0;
    g0[0] = 1u;                                   // count=1, user mode
    g0[1] = (unsigned)SM_STAGE;                   // lds_addr (buffer 0)
    g0[2] = (unsigned)(ga & 0xffffffffu);         // global_addr[31:0]
    g0[3] = (unsigned)((ga >> 32) & 0x01ffffffu)  // global_addr[56:32]
            | (2u << 30);                         // type = 2 ("image")
    v8i g1;
    g1[0] = 0x00010000;  // workgroup_mask=0, data_size=1 (2 bytes)
    g1[1] = (int)0x40000000u;  // tensor_dim0[15:0]=16384 in bits[63:48]
    g1[2] = 0x00010000;  // tensor_dim0[31:16]=0 | tensor_dim1[15:0]=1
    g1[3] = (int)0x40000000u;  // tensor_dim1[31:16]=0 | tile_dim0=16384
    g1[4] = 0x00000001;  // tile_dim1=1, tile_dim2=0
    g1[5] = 16384;       // tensor_dim0_stride low32
    g1[6] = 0;
    g1[7] = 0;
    v4i z4 = {0, 0, 0, 0};
    v8i z8 = {0, 0, 0, 0, 0, 0, 0, 0};
    // 6-arg toolchain variant: (g0, g1, g2, g3, g4, cpol)
    __builtin_amdgcn_tensor_load_to_lds(g0, g1, z4, z4, z8, 0);
    __builtin_amdgcn_s_wait_tensorcnt(0);
  }
#else
  {
    _Float16* st0 = (_Float16*)(smem + SM_STAGE);
    for (int i = tid; i < 128 * HDIM; i += 256) st0[i] = y0g[blockRow0 * HDIM + i];
  }
#endif

  // ---- stage combined weights into LDS, pre-swizzled to B-fragment order ----
  // B element (k,n) of tile (nt,kt): lane = n + 16*((k/8)&1), half idx = (k&7)+8*(k/16)
  for (int ww = tid; ww < 48 * 4 * 512; ww += 256) {
    int tile = ww >> 9;
    int r    = ww & 511;
    int ln   = r >> 4;
    int idx  = r & 15;
    int nt = tile >> 2, kt = tile & 3;
    int n = ln & 15;
    int k = (idx & 7) + 8 * (ln >> 4) + 16 * (idx >> 3);
    float v = (nt < 24) ? w_ih[(nt * 16 + n) * HDIM + kt * 32 + k]
                        : w_hh[((nt - 24) * 16 + n) * HDIM + kt * 32 + k];
    wl[ww] = (_Float16)v;
  }
  for (int i = tid; i < G3; i += 256) { bihL[i] = b_ih[i]; bhhL[i] = b_hh[i]; }
  for (int i = tid; i < HDIM; i += 256) owL[i] = out_w[i];
  __syncthreads();

  // SIRD state: rows owned by half-0 waves, one row per lane
  const int row = blockRow0 + slabRow0 + lane;
  float sS = 0.f, sI = 0.f, sR = 0.f, sD = 0.f, sN = 1.f;
  if (half == 0) {
    sS = sird[row * 5 + 0]; sI = sird[row * 5 + 1]; sR = sird[row * 5 + 2];
    sD = sird[row * 5 + 3]; sN = sird[row * 5 + 4];
  }
  const float outb = out_b[0];
  const float rcpN = __builtin_amdgcn_rcpf(sN);

  for (int t = 0; t < TSTEPS; ++t) {
    _Float16* stageR = (_Float16*)(smem + SM_STAGE + (t & 1) * 32768);
    _Float16* stageW = (_Float16*)(smem + SM_STAGE + ((t + 1) & 1) * 32768);

    // ---- gather A fragments of h(t) (x == h; at t==0 buffer0 holds y0) ----
    v16h a[4][2];
#pragma unroll
    for (int kt = 0; kt < 4; ++kt)
#pragma unroll
      for (int mi = 0; mi < 2; ++mi) {
        const _Float16* ap =
            stageR + (slabRow0 + mi * 16 + ln16) * HDIM + kt * 32 + 8 * g;
        v8h lo = *(const v8h*)ap;
        v8h hi = *(const v8h*)(ap + 16);
        a[kt][mi] = __builtin_shufflevector(
            lo, hi, 0, 1, 2, 3, 4, 5, 6, 7, 8, 9, 10, 11, 12, 13, 14, 15);
      }

    float pb[2][8];  // beta dot-product partials over this wave's u-tiles
#pragma unroll
    for (int mi = 0; mi < 2; ++mi)
#pragma unroll
      for (int v = 0; v < 8; ++v) pb[mi][v] = 0.0f;

    for (int j = 0; j < 4; ++j) {     // this wave's 4 u-tiles
      const int jt = half * 4 + j;
      const int nb = jt * 16 + ln16;
      const float bir = bihL[nb],       bhr = bhhL[nb];
      const float biz = bihL[128 + nb], bhz = bhhL[128 + nb];
      const float bin = bihL[256 + nb], bhn = bhhL[256 + nb];
      const float ow  = owL[nb];
      v8f accA[2], accB[2], r_[2], z_[2];

      // r gate: sigma(i_r + h_r) -> single accumulator, biases pre-summed
      // (at t==0, h==0 so gh == b_hh: skipping the H tiles is exact)
      gate_mma_pair(accA, a, wl, jt, 24 + jt, lane, bir + bhr, t != 0);
#pragma unroll
      for (int mi = 0; mi < 2; ++mi)
#pragma unroll
        for (int i = 0; i < 8; ++i) r_[mi][i] = fast_sigmoid(accA[mi][i]);

      // z gate: same trick
      gate_mma_pair(accA, a, wl, 8 + jt, 32 + jt, lane, biz + bhz, t != 0);
#pragma unroll
      for (int mi = 0; mi < 2; ++mi)
#pragma unroll
        for (int i = 0; i < 8; ++i) z_[mi][i] = fast_sigmoid(accA[mi][i]);

      // n gate: n = tanh(i_n + r * h_n) -> keep the two halves separate
      gate_mma_pair(accA, a, wl, 16 + jt, 0, lane, bin, false);
      if (t) gate_mma_pair(accB, a, wl, 40 + jt, 0, lane, bhn, false);
      else {
#pragma unroll
        for (int mi = 0; mi < 2; ++mi)
#pragma unroll
          for (int i = 0; i < 8; ++i) accB[mi][i] = bhn;
      }

      // h' = (1-z)*n + z*h, written to the WRITE buffer (double-buffered)
#pragma unroll
      for (int mi = 0; mi < 2; ++mi) {
#pragma unroll
        for (int v = 0; v < 8; ++v) {
          float nval = fast_tanh(accA[mi][v] + r_[mi][v] * accB[mi][v]);
          int m = slabRow0 + mi * 16 + 8 * g + v;          // C-layout row
          float hold = t ? (float)stageR[m * HDIM + nb] : 0.0f;
          float hnew = (1.0f - z_[mi][v]) * nval + z_[mi][v] * hold;
          stageW[m * HDIM + nb] = (_Float16)hnew;
          pb[mi][v] += hnew * ow;
        }
      }
    }  // j

    // partial beta dots: reduce over the 16-lane N group, publish per half
#pragma unroll
    for (int mi = 0; mi < 2; ++mi)
#pragma unroll
      for (int v = 0; v < 8; ++v) {
        float p = pb[mi][v];
        p += __shfl_xor(p, 1, 32);
        p += __shfl_xor(p, 2, 32);
        p += __shfl_xor(p, 4, 32);
        p += __shfl_xor(p, 8, 32);
        pb[mi][v] = p;
      }
    float* betaHalf = betaL + ((t & 1) * 2 + half) * 128;
    if (ln16 == 0) {
#pragma unroll
      for (int mi = 0; mi < 2; ++mi)
#pragma unroll
        for (int v = 0; v < 8; ++v)
          betaHalf[slabRow0 + mi * 16 + 8 * g + v] = pb[mi][v];
    }

    __syncthreads();  // h(t+1) complete + both beta halves published

    if (half == 0) {
      const float* bL = betaL + (t & 1) * 2 * 128;
      float be = fast_sigmoid(bL[slabRow0 + lane] + bL[128 + slabRow0 + lane] + outb);
      // SIRD step (elementwise, one row per lane)
      float si = sS * sI * be * rcpN;
      float ir = sI * 0.05f;   // RECOVER_RATE
      float dd = sI * 0.01f;   // DEATH_RATE
      sS -= si;
      sI += si - ir - dd;
      sR += ir;
      sD += dd;
    }
  }  // t

  if (half == 0) {
    out[row * 5 + 0] = sS;
    out[row * 5 + 1] = sI;
    out[row * 5 + 2] = sR;
    out[row * 5 + 3] = sD;
    out[row * 5 + 4] = sN;
  }
}

extern "C" void kernel_launch(void* const* d_in, const int* in_sizes, int n_in,
                              void* d_out, int out_size, void* d_ws, size_t ws_size,
                              hipStream_t stream) {
  (void)in_sizes; (void)n_in; (void)out_size; (void)ws_size;
  const float* hist  = (const float*)d_in[0];   // [B,60]
  const float* sird  = (const float*)d_in[1];   // [B,5]
  const float* enc_w = (const float*)d_in[2];   // [128,60]
  const float* enc_b = (const float*)d_in[3];   // [128]
  const float* w_ih  = (const float*)d_in[4];   // [384,128]
  const float* w_hh  = (const float*)d_in[5];   // [384,128]
  const float* b_ih  = (const float*)d_in[6];   // [384]
  const float* b_hh  = (const float*)d_in[7];   // [384]
  const float* out_w = (const float*)d_in[8];   // [128]
  const float* out_b = (const float*)d_in[9];   // [1]
  float* out = (float*)d_out;                   // [B,5]
  _Float16* y0 = (_Float16*)d_ws;               // [B,128] f16 scratch (1 MB)

  enc_kernel_42880953483646<<<(BATCH * HDIM) / 256, 256, 0, stream>>>(
      hist, enc_w, enc_b, y0);
  gru_sird_kernel_42880953483646<<<BATCH / 128, 256, SM_TOTAL, stream>>>(
      y0, sird, w_ih, w_hh, b_ih, b_hh, out_w, out_b, out);
}
